// FovealProcessor_45518063403228
// MI455X (gfx1250) — matrix operand order repaired and checked by reading, not compiled, not run on hardware
//
#include <hip/hip_runtime.h>

// ---------------------------------------------------------------------------
// FovealProcessor for MI455X (gfx1250, wave32, WMMA)
// Dataflow insight: outputs are (cls_out, new_acc). new_acc needs no
// attention; cls_out needs attention for QUERY ROW 0 only. The single
// compute-dominant stage is the K/V projection GEMM
// (16 x 769x1024 @ 1024x2048 = 51.7 GFLOP) which runs as a bf16 WMMA GEMM
// (v_wmma_f32_16x16x32_bf16) with double-buffered LDS fed by
// global_load_async_to_lds_b128 (ASYNCcnt pipeline). Everything else is
// memory-bound at M=16 and stays fp32 VALU.
// ---------------------------------------------------------------------------

#define DIMD    1024
#define NSEQ    4096
#define BATCH   16
#define ACCROWS 512
#define WIN     256
#define LROWS   769          // 1 (cls) + 512 (acc) + 256 (window)
#define MP      832          // LROWS padded to 13 * 64 for GEMM M-tiling
#define KV_N    2048         // K and V projections

typedef __attribute__((ext_vector_type(16))) __bf16 bf16x16;
typedef __attribute__((ext_vector_type(8)))  float  v8f;

union Frag16 { bf16x16 v; uint4 u[2]; };

__device__ __forceinline__ float bf2f(unsigned short h) {
    union { unsigned int u; float f; } c; c.u = ((unsigned int)h) << 16; return c.f;
}
__device__ __forceinline__ unsigned short f2bf(float f) {
    union { float f; unsigned int u; } c; c.f = f;
    unsigned int u = c.u;
    unsigned int r = u + 0x7FFFu + ((u >> 16) & 1u);   // round-to-nearest-even
    return (unsigned short)(r >> 16);
}

// Async memory -> LDS copy, 16 B per lane (VGLOBAL async op, ASYNCcnt).
// VDST holds the wave-relative LDS byte address: for a flat shared pointer
// the SHARED_BASE aperture lives in addr[63:32], so the low 32 bits are the
// LDS byte offset (ISA 10.2 aperture calculation).
__device__ __forceinline__ void async_copy_b128(const void* gptr, void* lptr) {
    unsigned lds = (unsigned)(uintptr_t)lptr;
    asm volatile("global_load_async_to_lds_b128 %0, %1, off"
                 :: "v"(lds), "v"(gptr) : "memory");
}
__device__ __forceinline__ void wait_async0() {
    asm volatile("s_wait_asynccnt 0" ::: "memory");
}

// ---------------------------------------------------------------------------
// K1: gather ctx rows (state | acc_ctx | window), write new_acc part of d_out,
//     layernorm each row, emit bf16 "normed" (rows >= LROWS zero-padded).
// grid: (MP, BATCH), block: 256
// ---------------------------------------------------------------------------
__global__ __launch_bounds__(256) void build_ctx_norm(
    const float* __restrict__ x, const float* __restrict__ state,
    const float* __restrict__ acc_ctx, const int* __restrict__ fix_pt,
    const float* __restrict__ g, const float* __restrict__ beta,
    unsigned short* __restrict__ normed, float* __restrict__ out)
{
    const int l = blockIdx.x, b = blockIdx.y, tid = threadIdx.x;
    __shared__ float r1[256], r2[256];
    unsigned short* nrow = normed + ((size_t)b * MP + l) * DIMD;

    if (l >= LROWS) {                 // zero pad rows for boundless GEMM loads
        for (int i = tid; i < DIMD; i += 256) nrow[i] = 0;
        return;
    }
    const float* src;
    if (l == 0)        src = state + (size_t)b * DIMD;
    else if (l <= ACCROWS) src = acc_ctx + ((size_t)b * ACCROWS + (l - 1)) * DIMD;
    else {
        int start = fix_pt[b] - (WIN / 2);
        if (start < 0) start = 0;
        if (start > NSEQ - WIN) start = NSEQ - WIN;
        src = x + ((size_t)b * NSEQ + start + (l - 1 - ACCROWS)) * DIMD;
    }
    float v[4], s = 0.f, s2 = 0.f;
    for (int j = 0; j < 4; ++j) {
        float t = src[tid + j * 256];
        v[j] = t; s += t; s2 += t * t;
    }
    if (l >= 1) {                     // new_acc = concat(acc_ctx, window)
        float* dst = out + BATCH * DIMD + ((size_t)b * (LROWS - 1) + (l - 1)) * DIMD;
        for (int j = 0; j < 4; ++j) dst[tid + j * 256] = v[j];
    }
    r1[tid] = s; r2[tid] = s2; __syncthreads();
    for (int off = 128; off > 0; off >>= 1) {
        if (tid < off) { r1[tid] += r1[tid + off]; r2[tid] += r2[tid + off]; }
        __syncthreads();
    }
    float mean = r1[0] * (1.0f / DIMD);
    float var  = r2[0] * (1.0f / DIMD) - mean * mean;
    float rstd = rsqrtf(var + 1e-5f);
    for (int j = 0; j < 4; ++j) {
        int i = tid + j * 256;
        nrow[i] = f2bf((v[j] - mean) * rstd * g[i] + beta[i]);
    }
}

// ---------------------------------------------------------------------------
// K0: convert K/V weight rows (in_proj_w rows 1024..3071) fp32 -> bf16
// ---------------------------------------------------------------------------
__global__ __launch_bounds__(256) void conv_wkv(
    const float* __restrict__ w, unsigned short* __restrict__ wkv)
{
    size_t i = (size_t)blockIdx.x * 256 + threadIdx.x;   // 2048*1024 total
    wkv[i] = f2bf(w[(size_t)DIMD * DIMD + i]);
}

// ---------------------------------------------------------------------------
// K2: KV GEMM (WMMA). C[b,m,n] = normed[b,m,:] . Wkv[n,:] + bias
// A: [MP x 1024] bf16 row-major, B: [2048 x 1024] bf16 row-major (K contig).
// 256 thr = 8 waves; tile 64(M) x 128(N); K-step 32; double-buffered LDS
// filled by global_load_async_to_lds_b128 overlapped with WMMA compute.
// grid: (2048/128, MP/64, BATCH)
// ---------------------------------------------------------------------------
#define LDSK 48   // padded K stride (bf16 units): 96 B rows keep b128 aligned
__global__ __launch_bounds__(256) void kv_gemm(
    const unsigned short* __restrict__ normed,
    const unsigned short* __restrict__ wkv,
    const float* __restrict__ in_proj_b,
    unsigned short* __restrict__ kv)
{
    __shared__ __align__(16) unsigned short As[2][64 * LDSK];
    __shared__ __align__(16) unsigned short Bs[2][128 * LDSK];

    const int b    = blockIdx.z;
    const int m0   = blockIdx.y * 64;
    const int n0   = blockIdx.x * 128;
    const int tid  = threadIdx.x;
    const int lane = tid & 31;
    const int wave = tid >> 5;
    const int msub = (wave & 3) * 16;     // 4 waves across M
    const int nsub = (wave >> 2) * 64;    // 2 waves across N, 4 N-tiles each

    // copy tasks: A is 64x32 (1 x b128/thread), B is 128x32 (2 x b128/thread)
    const int r4 = tid >> 2;              // 0..63
    const int c8 = (tid & 3) * 8;         // 0,8,16,24 (bf16 units)

    const unsigned short* gAb = normed + (size_t)b * MP * DIMD;

    v8f z = {0,0,0,0,0,0,0,0};
    v8f acc[4]; acc[0] = z; acc[1] = z; acc[2] = z; acc[3] = z;

    // stage K-tile k0 into LDS buffer q (async, no VGPR round-trip)
    auto stage = [&](int k0, int q) {
        async_copy_b128(gAb + (size_t)(m0 + r4) * DIMD + k0 + c8,
                        &As[q][r4 * LDSK + c8]);
        const unsigned short* gB0 = wkv + (size_t)(n0 + r4) * DIMD + k0 + c8;
        async_copy_b128(gB0,                     &Bs[q][r4 * LDSK + c8]);
        async_copy_b128(gB0 + (size_t)64 * DIMD, &Bs[q][(r4 + 64) * LDSK + c8]);
    };

    stage(0, 0);
    wait_async0();
    __syncthreads();

    const int lrow = msub + (lane & 15);
    const int klo  = (lane < 16) ? 0 : 16;
    const int khi  = (lane < 16) ? 8 : 24;
    const int bko  = (lane < 16) ? 0 : 16;

    int p = 0;
    for (int k0 = 0; k0 < DIMD; k0 += 32) {
        const bool has_next = (k0 + 32) < DIMD;
        if (has_next) stage(k0 + 32, p ^ 1);   // overlap DMA with compute

        // preload A fragment + ALL four B fragments, then 4 back-to-back WMMAs
        const unsigned short* Ab = &As[p][0];
        const unsigned short* Bb = &Bs[p][0];
        Frag16 a, bf[4];
        a.u[0] = *reinterpret_cast<const uint4*>(Ab + lrow * LDSK + klo);
        a.u[1] = *reinterpret_cast<const uint4*>(Ab + lrow * LDSK + khi);
        #pragma unroll
        for (int t = 0; t < 4; ++t) {
            int col = nsub + t * 16 + (lane & 15);
            bf[t].u[0] = *reinterpret_cast<const uint4*>(Bb + col * LDSK + bko);
            bf[t].u[1] = *reinterpret_cast<const uint4*>(Bb + col * LDSK + bko + 8);
        }
        #pragma unroll
        for (int t = 0; t < 4; ++t)
            acc[t] = __builtin_amdgcn_wmma_f32_16x16x32_bf16(
                false, a.v, false, bf[t].v, (short)0, acc[t], false, false);

        if (has_next) {
            wait_async0();        // own async writes landed in LDS
            __syncthreads();      // everyone's writes published
        }
        p ^= 1;
    }

    // Epilogue: C layout — VGPR r: M = r + 8*(lane>=16), N = lane%16
    const int mbase = m0 + msub + ((lane >> 4) << 3);
    #pragma unroll
    for (int t = 0; t < 4; ++t) {
        int n = n0 + nsub + t * 16 + (lane & 15);
        float bias = in_proj_b[DIMD + n];
        #pragma unroll
        for (int r = 0; r < 8; ++r) {
            int m = mbase + r;
            if (m < LROWS)
                kv[((size_t)b * LROWS + m) * KV_N + n] = f2bf(acc[t][r] + bias);
        }
    }
}

// ---------------------------------------------------------------------------
// K3: q projection for row 0 only: q0[b][j] = normed[b,0,:] . Wq[j,:] + bias
// ---------------------------------------------------------------------------
__global__ __launch_bounds__(256) void q_proj(
    const unsigned short* __restrict__ normed, const float* __restrict__ W,
    const float* __restrict__ bias, float* __restrict__ q0)
{
    int idx = blockIdx.x * 256 + threadIdx.x;   // 16 * 1024
    int b = idx >> 10, j = idx & 1023;
    const unsigned short* nr = normed + (size_t)b * MP * DIMD;  // row 0
    const float* wr = W + (size_t)j * DIMD;
    float acc = bias[j];
    for (int d = 0; d < DIMD; ++d) acc += bf2f(nr[d]) * wr[d];
    q0[idx] = acc;
}

// ---------------------------------------------------------------------------
// K4: attention for the single query row, per (b, h). Softmax over 769 keys.
// grid: B*H = 256 blocks, 256 threads.
// ---------------------------------------------------------------------------
__global__ __launch_bounds__(256) void attn_row0(
    const float* __restrict__ q0, const unsigned short* __restrict__ kv,
    float* __restrict__ attn_o)
{
    const int b = blockIdx.x >> 4, h = blockIdx.x & 15, tid = threadIdx.x;
    __shared__ float qs[64];
    __shared__ float sc[LROWS];
    __shared__ float red[256];

    if (tid < 64) qs[tid] = q0[b * DIMD + h * 64 + tid] * 0.125f;  // 1/sqrt(64)
    __syncthreads();

    float lmax = -1e30f;
    for (int l = tid; l < LROWS; l += 256) {
        const unsigned short* kr = kv + ((size_t)b * LROWS + l) * KV_N + h * 64;
        float s = 0.f;
        for (int d = 0; d < 64; ++d) s += qs[d] * bf2f(kr[d]);
        sc[l] = s;
        lmax = fmaxf(lmax, s);
    }
    red[tid] = lmax; __syncthreads();
    for (int off = 128; off > 0; off >>= 1) {
        if (tid < off) red[tid] = fmaxf(red[tid], red[tid + off]);
        __syncthreads();
    }
    float m = red[0]; __syncthreads();

    float lsum = 0.f;
    for (int l = tid; l < LROWS; l += 256) {
        float p = expf(sc[l] - m); sc[l] = p; lsum += p;
    }
    red[tid] = lsum; __syncthreads();
    for (int off = 128; off > 0; off >>= 1) {
        if (tid < off) red[tid] += red[tid + off];
        __syncthreads();
    }
    float denom = red[0]; __syncthreads();

    // o[d] = sum_l p_l * v[l][d] ; 64 dims x 4 l-chunks
    const int d = tid & 63, chunk = tid >> 6;
    float acc = 0.f;
    for (int l = chunk; l < LROWS; l += 4)
        acc += sc[l] * bf2f(kv[((size_t)b * LROWS + l) * KV_N + DIMD + h * 64 + d]);
    red[tid] = acc; __syncthreads();
    if (tid < 64)
        attn_o[b * DIMD + h * 64 + tid] =
            (red[tid] + red[64 + tid] + red[128 + tid] + red[192 + tid]) / denom;
}

// ---------------------------------------------------------------------------
// K5: out projection + residual (cls_pre = state + o @ out_w.T + out_b)
// ---------------------------------------------------------------------------
__global__ __launch_bounds__(256) void out_proj(
    const float* __restrict__ attn_o, const float* __restrict__ out_w,
    const float* __restrict__ out_b, const float* __restrict__ state,
    float* __restrict__ cls_pre)
{
    int idx = blockIdx.x * 256 + threadIdx.x;  // 16 * 1024
    int b = idx >> 10, i = idx & 1023;
    const float* o  = attn_o + (size_t)b * DIMD;
    const float* wr = out_w + (size_t)i * DIMD;
    float acc = out_b[i] + state[idx];
    for (int j = 0; j < DIMD; ++j) acc += o[j] * wr[j];
    cls_pre[idx] = acc;
}

// ---------------------------------------------------------------------------
// K6: layernorm2 on cls rows (16 rows)
// ---------------------------------------------------------------------------
__global__ __launch_bounds__(256) void ln2_rows(
    const float* __restrict__ cls_pre, const float* __restrict__ g,
    const float* __restrict__ beta, float* __restrict__ h_in)
{
    const int b = blockIdx.x, tid = threadIdx.x;
    __shared__ float r1[256], r2[256];
    const float* src = cls_pre + (size_t)b * DIMD;
    float v[4], s = 0.f, s2 = 0.f;
    for (int j = 0; j < 4; ++j) {
        float t = src[tid + j * 256]; v[j] = t; s += t; s2 += t * t;
    }
    r1[tid] = s; r2[tid] = s2; __syncthreads();
    for (int off = 128; off > 0; off >>= 1) {
        if (tid < off) { r1[tid] += r1[tid + off]; r2[tid] += r2[tid + off]; }
        __syncthreads();
    }
    float mean = r1[0] * (1.0f / DIMD);
    float var  = r2[0] * (1.0f / DIMD) - mean * mean;
    float rstd = rsqrtf(var + 1e-5f);
    for (int j = 0; j < 4; ++j) {
        int i = tid + j * 256;
        h_in[(size_t)b * DIMD + i] = (v[j] - mean) * rstd * g[i] + beta[i];
    }
}

// ---------------------------------------------------------------------------
// K7: MLP layer 1 + exact GELU.  w1 is (D, 4D): column access, coalesced in j.
// ---------------------------------------------------------------------------
__global__ __launch_bounds__(256) void mlp1(
    const float* __restrict__ h_in, const float* __restrict__ w1,
    const float* __restrict__ b1, float* __restrict__ h)
{
    int idx = blockIdx.x * 256 + threadIdx.x;  // 16 * 4096
    int b = idx >> 12, j = idx & 4095;
    const float* xr = h_in + (size_t)b * DIMD;
    float acc = b1[j];
    for (int i = 0; i < DIMD; ++i) acc += xr[i] * w1[(size_t)i * (4 * DIMD) + j];
    h[idx] = 0.5f * acc * (1.0f + erff(acc * 0.70710678118654752f));
}

// ---------------------------------------------------------------------------
// K8: MLP layer 2 + residual -> cls_out (start of d_out)
// ---------------------------------------------------------------------------
__global__ __launch_bounds__(256) void mlp2(
    const float* __restrict__ cls_pre, const float* __restrict__ h,
    const float* __restrict__ w2, const float* __restrict__ b2,
    float* __restrict__ out)
{
    int idx = blockIdx.x * 256 + threadIdx.x;  // 16 * 1024
    int b = idx >> 10, i = idx & 1023;
    const float* hr = h + (size_t)b * (4 * DIMD);
    float acc = b2[i] + cls_pre[idx];
    for (int j = 0; j < 4 * DIMD; ++j) acc += hr[j] * w2[(size_t)j * DIMD + i];
    out[idx] = acc;
}

// ---------------------------------------------------------------------------
extern "C" void kernel_launch(void* const* d_in, const int* in_sizes, int n_in,
                              void* d_out, int out_size, void* d_ws, size_t ws_size,
                              hipStream_t stream)
{
    const float* x         = (const float*)d_in[0];
    const float* state     = (const float*)d_in[1];
    const float* acc_ctx   = (const float*)d_in[2];
    const int*   fix_pt    = (const int*)  d_in[3];
    const float* in_proj_w = (const float*)d_in[4];
    const float* in_proj_b = (const float*)d_in[5];
    const float* out_w     = (const float*)d_in[6];
    const float* out_b     = (const float*)d_in[7];
    const float* ln1_g     = (const float*)d_in[8];
    const float* ln1_b     = (const float*)d_in[9];
    const float* ln2_g     = (const float*)d_in[10];
    const float* ln2_b     = (const float*)d_in[11];
    const float* w1        = (const float*)d_in[12];
    const float* b1        = (const float*)d_in[13];
    const float* w2        = (const float*)d_in[14];
    const float* b2        = (const float*)d_in[15];
    float* out = (float*)d_out;

    // workspace layout (bytes)
    char* ws = (char*)d_ws;
    size_t off = 0;
    unsigned short* normed = (unsigned short*)(ws + off); off += (size_t)BATCH * MP * DIMD * 2;      // 27.3 MB
    unsigned short* kvbuf  = (unsigned short*)(ws + off); off += (size_t)BATCH * LROWS * KV_N * 2;   // 50.4 MB
    unsigned short* wkv    = (unsigned short*)(ws + off); off += (size_t)KV_N * DIMD * 2;            //  4.2 MB
    float* q0      = (float*)(ws + off); off += (size_t)BATCH * DIMD * 4;
    float* attn_o  = (float*)(ws + off); off += (size_t)BATCH * DIMD * 4;
    float* cls_pre = (float*)(ws + off); off += (size_t)BATCH * DIMD * 4;
    float* h_in    = (float*)(ws + off); off += (size_t)BATCH * DIMD * 4;
    float* hbuf    = (float*)(ws + off); off += (size_t)BATCH * 4 * DIMD * 4;

    conv_wkv<<<dim3((KV_N * DIMD) / 256), 256, 0, stream>>>(in_proj_w, wkv);
    build_ctx_norm<<<dim3(MP, BATCH), 256, 0, stream>>>(
        x, state, acc_ctx, fix_pt, ln1_g, ln1_b, normed, out);
    kv_gemm<<<dim3(KV_N / 128, MP / 64, BATCH), 256, 0, stream>>>(
        normed, wkv, in_proj_b, kvbuf);
    q_proj<<<dim3((BATCH * DIMD) / 256), 256, 0, stream>>>(normed, in_proj_w, in_proj_b, q0);
    attn_row0<<<dim3(BATCH * 16), 256, 0, stream>>>(q0, kvbuf, attn_o);
    out_proj<<<dim3((BATCH * DIMD) / 256), 256, 0, stream>>>(attn_o, out_w, out_b, state, cls_pre);
    ln2_rows<<<dim3(BATCH), 256, 0, stream>>>(cls_pre, ln2_g, ln2_b, h_in);
    mlp1<<<dim3((BATCH * 4 * DIMD) / 256), 256, 0, stream>>>(h_in, w1, b1, hbuf);
    mlp2<<<dim3((BATCH * DIMD) / 256), 256, 0, stream>>>(cls_pre, hbuf, w2, b2, out);
}
